// AttentionLayer_88270167868124
// MI455X (gfx1250) — matrix operand order repaired and checked by reading, not compiled, not run on hardware
//
#include <hip/hip_runtime.h>
#include <math.h>

typedef __attribute__((ext_vector_type(16))) _Float16 v16h;
typedef __attribute__((ext_vector_type(8)))  _Float16 v8h;
typedef __attribute__((ext_vector_type(8)))  float    v8f;

#define EDIM 128
#define ADIM 64
#define SEQ  200
#define STILES 13      // ceil(200/16)
#define XSTRIDE 136    // halves per X-tile row: 272B -> 16B aligned chunks, bank-quad spread
#define BSLOT 24       // halves per lane per B-fragment slot (16 used + 8 pad): 48B, 16B aligned

// ---------------------------------------------------------------------------
// Kernel 0: fold W1's four 128x64 blocks into the restructured operators:
//   Ap = W1a + W1d   (coefficient of x)
//   Cp = W1c         (coefficient of x*t)
//   Bm = W1b - W1d   (coefficient of t -> per-batch bias u)
// ---------------------------------------------------------------------------
__global__ __launch_bounds__(256)
void prep_kernel(const float* __restrict__ W1, float* __restrict__ ws) {
  int idx = blockIdx.x * 256 + threadIdx.x;
  if (idx < EDIM * ADIM) {
    float w0 = W1[idx];                    // block 0: behaviors
    float w1 = W1[EDIM * ADIM + idx];      // block 1: target
    float w2 = W1[2 * EDIM * ADIM + idx];  // block 2: behaviors * t
    float w3 = W1[3 * EDIM * ADIM + idx];  // block 3: behaviors - t
    ws[idx]                   = w0 + w3;   // Ap
    ws[EDIM * ADIM + idx]     = w2;        // Cp
    ws[2 * EDIM * ADIM + idx] = w1 - w3;   // Bm
  }
}

// ---------------------------------------------------------------------------
// Main kernel: one block per batch, 256 threads = 8 waves (wave32).
// ---------------------------------------------------------------------------
__global__ __launch_bounds__(256)
void attn_kernel(const float* __restrict__ behaviors,
                 const float* __restrict__ target,
                 const float* __restrict__ b1,
                 const float* __restrict__ W2,
                 const float* __restrict__ b2,
                 const float* __restrict__ ws,
                 float* __restrict__ out) {
  __shared__ float    tsh[EDIM];
  __shared__ float    ush[ADIM];
  __shared__ float    outacc[EDIM];
  // M(b) pre-swizzled into WMMA B-operand order: [frag(kk,nt)][lane][BSLOT halves]
  __shared__ _Float16 Mhsw[16 * 32 * BSLOT];
  __shared__ _Float16 Xh[8][16 * XSTRIDE];        // per-wave f16 X tile

  const int tid  = threadIdx.x;
  const int b    = blockIdx.x;
  const int wave = tid >> 5;
  const int lane = tid & 31;
  const int nl   = lane & 15;
  const int hi   = lane >> 4;                     // 0 or 1 (half-wave)

  const float* Ap = ws;
  const float* Cp = ws + EDIM * ADIM;
  const float* Bm = ws + 2 * EDIM * ADIM;

  if (tid < EDIM) { tsh[tid] = target[(size_t)b * EDIM + tid]; outacc[tid] = 0.f; }
  __syncthreads();

  // Build M(b) = Ap + diag(t) * Cp, storing each element straight into its
  // B-fragment position: K = e = kk*32 + hi*16 + h, N = a = nt*16 + nl.
  for (int i = 0; i < (EDIM * ADIM) / 256; ++i) {
    int idx = tid + i * 256;              // idx = e*64 + a
    int e = idx >> 6, a = idx & 63;
    float v = Ap[idx] + tsh[e] * Cp[idx];
    int kk = e >> 5, kr = e & 31;
    int frag = kk * 4 + (a >> 4);
    int ln   = (kr >> 4) * 16 + (a & 15);
    int h    = kr & 15;
    Mhsw[(frag * 32 + ln) * BSLOT + h] = (_Float16)v;
  }
  // u(b) = b1 + t @ Bm
  if (tid < ADIM) {
    float acc = b1[tid];
    for (int e = 0; e < EDIM; ++e) acc += tsh[e] * Bm[e * ADIM + tid];
    ush[tid] = acc;
  }
  __syncthreads();

  // per-lane W2 slice and scalar b2
  float w2v[4];
  #pragma unroll
  for (int nt = 0; nt < 4; ++nt) w2v[nt] = W2[nt * 16 + nl];
  const float b2s = b2[0];

  float wacc[4] = {0.f, 0.f, 0.f, 0.f};
  _Float16* xt = &Xh[wave][0];

  for (int st = wave; st < STILES; st += 8) {
    const int s0 = st * 16;

    // Stage 16x128 fp32 rows -> f16 LDS tile (coalesced float4 loads, zero-pad s>=200)
    for (int r = 0; r < 16; ++r) {
      int s = s0 + r;
      float4 xv = make_float4(0.f, 0.f, 0.f, 0.f);
      if (s < SEQ)
        xv = *(const float4*)(behaviors + ((size_t)b * SEQ + s) * EDIM + lane * 4);
      _Float16* dst = xt + r * XSTRIDE + lane * 4;
      dst[0] = (_Float16)xv.x; dst[1] = (_Float16)xv.y;
      dst[2] = (_Float16)xv.z; dst[3] = (_Float16)xv.w;
    }
    __builtin_amdgcn_wave_barrier();   // same-wave LDS RAW ordering fence

    v8f cacc[4];
    #pragma unroll
    for (int nt = 0; nt < 4; ++nt) cacc[nt] = (v8f){0.f,0.f,0.f,0.f,0.f,0.f,0.f,0.f};

    const _Float16* xrow = xt + nl * XSTRIDE;  // row M = nl
    const int akoff = hi * 16;                 // half-wave K offset for A operand

    #pragma unroll
    for (int kk = 0; kk < 4; ++kk) {
      // A fragment: halves = xrow[base..base+15]
      // -> two ds_load_b128, no element shuffling
      const v8h* pa = (const v8h*)(xrow + kk * 32 + akoff);
      v8h alo = pa[0];
      v8h ahi = pa[1];
      v16h a = __builtin_shufflevector(alo, ahi,
                 0,1,2,3,4,5,6,7, 8,9,10,11,12,13,14,15);
      #pragma unroll
      for (int nt = 0; nt < 4; ++nt) {
        // B fragment: one contiguous 32B per-lane load (2x ds_load_b128)
        v16h bfrag = *(const v16h*)(&Mhsw[((kk * 4 + nt) * 32 + lane) * BSLOT]);
        cacc[nt] = __builtin_amdgcn_wmma_f32_16x16x32_f16(
            false, a, false, bfrag, (short)0, cacc[nt], false, false);
      }
    }

    // Epilogue: relu(+u), dot with W2 across N (xor-shuffle reduce over 16 lanes)
    float lr[8];
    #pragma unroll
    for (int r = 0; r < 8; ++r) lr[r] = 0.f;
    #pragma unroll
    for (int nt = 0; nt < 4; ++nt) {
      float uu = ush[nt * 16 + nl];
      #pragma unroll
      for (int r = 0; r < 8; ++r) {
        float h = cacc[nt][r] + uu;
        h = h > 0.f ? h : 0.f;
        lr[r] += h * w2v[nt];
      }
    }
    #pragma unroll
    for (int r = 0; r < 8; ++r) {
      lr[r] += __shfl_xor(lr[r], 1, 32);
      lr[r] += __shfl_xor(lr[r], 2, 32);
      lr[r] += __shfl_xor(lr[r], 4, 32);
      lr[r] += __shfl_xor(lr[r], 8, 32);
      int s = s0 + hi * 8 + r;               // C/D layout: lane<16 -> M=r, else M=r+8
      float z = lr[r] + b2s;
      // sigmoid via native v_exp_f32 / v_rcp_f32
      float e = __builtin_amdgcn_exp2f(z * -1.4426950408889634f);
      lr[r] = (s < SEQ) ? __builtin_amdgcn_rcpf(1.f + e) : 0.f;
    }
    // Broadcast the 16 per-row weights to all lanes
    float wall[16];
    #pragma unroll
    for (int m = 0; m < 8; ++m) {
      wall[m]     = __shfl(lr[m], 0, 32);
      wall[m + 8] = __shfl(lr[m], 16, 32);
    }
    // out partial: each lane owns columns [4*lane, 4*lane+4)
    #pragma unroll
    for (int r = 0; r < 16; ++r) {
      float ww = wall[r];
      const _Float16* xr = xt + r * XSTRIDE + lane * 4;
      wacc[0] += ww * (float)xr[0];
      wacc[1] += ww * (float)xr[1];
      wacc[2] += ww * (float)xr[2];
      wacc[3] += ww * (float)xr[3];
    }
    __builtin_amdgcn_wave_barrier();   // WAR fence before next tile overwrites Xh
  }

  // Combine waves via LDS float atomics (ds_add_f32), then store
  #pragma unroll
  for (int j = 0; j < 4; ++j) atomicAdd(&outacc[lane * 4 + j], wacc[j]);
  __syncthreads();
  if (tid < EDIM) out[(size_t)b * EDIM + tid] = outacc[tid];
}

// ---------------------------------------------------------------------------
extern "C" void kernel_launch(void* const* d_in, const int* in_sizes, int n_in,
                              void* d_out, int out_size, void* d_ws, size_t ws_size,
                              hipStream_t stream) {
  const float* behaviors = (const float*)d_in[0];
  const float* target    = (const float*)d_in[1];
  const float* W1        = (const float*)d_in[2];
  const float* b1        = (const float*)d_in[3];
  const float* W2        = (const float*)d_in[4];
  const float* b2        = (const float*)d_in[5];
  float* out = (float*)d_out;
  float* ws  = (float*)d_ws;    // needs 3 * 128*64 * 4 = 96 KB

  int B = in_sizes[1] / EDIM;   // target is (B, 128)

  prep_kernel<<<(EDIM * ADIM + 255) / 256, 256, 0, stream>>>(W1, ws);
  attn_kernel<<<B, 256, 0, stream>>>(behaviors, target, b1, W2, b2, ws, out);
}
